// HSDGNN_block_47278999994425
// MI455X (gfx1250) — matrix operand order repaired and checked by reading, not compiled, not run on hardware
//
#include <hip/hip_runtime.h>
#include <hip/hip_bf16.h>

#define B_  64
#define T_  12
#define N_  307
#define I_  3
#define R_  64
#define D_  10
#define BN_ (B_ * N_)   /* 19648 = 16*1228 */
#define NP_ 320         /* N padded to 16 multiple */
#define TB_ (T_ * B_)   /* 768 = 16*48 */

typedef __attribute__((ext_vector_type(16))) _Float16 v16h;
typedef __attribute__((ext_vector_type(8)))  _Float16 v8h;
typedef __attribute__((ext_vector_type(8)))  float    v8f;

__device__ __forceinline__ float sigm(float x) { return 1.0f / (1.0f + __expf(-x)); }

__device__ __forceinline__ v8f zero8f() {
    v8f r;
#pragma unroll
    for (int i = 0; i < 8; ++i) r[i] = 0.0f;
    return r;
}
__device__ __forceinline__ v8h zero8h() {
    v8h r;
#pragma unroll
    for (int i = 0; i < 8; ++i) r[i] = (_Float16)0.0f;
    return r;
}
__device__ __forceinline__ v16h zero16h() {
    v16h r;
#pragma unroll
    for (int i = 0; i < 16; ++i) r[i] = (_Float16)0.0f;
    return r;
}
__device__ __forceinline__ v16h cat16(v8h lo, v8h hi) {
    v16h r;
#pragma unroll
    for (int i = 0; i < 8; ++i) { r[i] = lo[i]; r[i + 8] = hi[i]; }
    return r;
}
__device__ __forceinline__ v8h  ld8(const _Float16* p)  { return *(const v8h*)p; }
__device__ __forceinline__ v16h ld16(const _Float16* p) { return *(const v16h*)p; }

#define WMMA_F16(a, b, c) \
    __builtin_amdgcn_wmma_f32_16x16x32_f16(false, (a), false, (b), (short)0, (c), false, false)

// ---------------------------------------------------------------------------
// Zero-init hidden states
// ---------------------------------------------------------------------------
__global__ void k_zero_h(_Float16* h1, _Float16* h2) {
    int i = blockIdx.x * blockDim.x + threadIdx.x;
    if (i < BN_ * R_) { h1[i] = (_Float16)0.0f; h2[i] = (_Float16)0.0f; }
}

// ---------------------------------------------------------------------------
// Pack weight matrices into WMMA B-operand panels (f16).
// Panel element: dst[(tile*32 + lane)*16 + j] = W[col][k]
//   col = ct*16 + lane%16 ; k = kt*32 + 16*(j/8) + 8*(lane/16) + j%8  (zero beyond K)
// tiles: [0,24) whh1  [24,36) wih1(K=10)  [36,60) whh2  [60,84) wih2  [84,92) dw1
// ---------------------------------------------------------------------------
__global__ void k_pack(const float* whh1, const float* wih1, const float* whh2,
                       const float* wih2, const float* dw,
                       _Float16* pWhh1, _Float16* pWih1, _Float16* pWhh2,
                       _Float16* pWih2, _Float16* pDw1) {
    int tid = blockIdx.x * blockDim.x + threadIdx.x;
    int tile = tid >> 9;
    int e = tid & 511;
    int lane = e >> 4, j = e & 15;
    const float* src; _Float16* dst; int K, ldk, ct, kt, lt;
    if (tile < 24)      { src = whh1; dst = pWhh1; lt = tile;      ct = lt >> 1; kt = lt & 1; K = 64; ldk = 64; }
    else if (tile < 36) { src = wih1; dst = pWih1; lt = tile - 24; ct = lt;      kt = 0;      K = 10; ldk = 10; }
    else if (tile < 60) { src = whh2; dst = pWhh2; lt = tile - 36; ct = lt >> 1; kt = lt & 1; K = 64; ldk = 64; }
    else if (tile < 84) { src = wih2; dst = pWih2; lt = tile - 60; ct = lt >> 1; kt = lt & 1; K = 64; ldk = 64; }
    else                { src = dw;   dst = pDw1;  lt = tile - 84; ct = lt >> 1; kt = lt & 1; K = 64; ldk = 128; }
    int col = ct * 16 + (lane & 15);
    int k = kt * 32 + ((j >> 3) << 4) + ((lane >> 4) << 3) + (j & 7);
    float v = (k < K) ? src[col * ldk + k] : 0.0f;
    dst[(lt * 32 + lane) * 16 + j] = (_Float16)v;
}

// ---------------------------------------------------------------------------
// Per-node adapted weights folded with conv half:
//   W'[n][c][q] = sum_r (sum_d emb2[n,d] pool[d,c,r]) * diff_w[q][64+r]   (panel f16)
//   bias2[n][q] = sum_r (emb2[n]·bias_pool[:,r])      * diff_w[q][64+r] + diff_b[q]
// ---------------------------------------------------------------------------
__global__ void k_prep_n(const float* __restrict__ emb2, const float* __restrict__ pool,
                         const float* __restrict__ bias_pool, const float* __restrict__ diff_w,
                         const float* __restrict__ diff_b,
                         _Float16* __restrict__ pWn, float* __restrict__ bias2) {
    __shared__ float tW[64 * 64];
    __shared__ float tB[64];
    int n = blockIdx.x;
    float ed[D_];
#pragma unroll
    for (int d = 0; d < D_; ++d) ed[d] = emb2[n * D_ + d];
    for (int i = threadIdx.x; i < 4096; i += blockDim.x) {
        int c = i >> 6, r = i & 63;
        float s = 0.0f;
#pragma unroll
        for (int d = 0; d < D_; ++d) s += ed[d] * pool[(d * 64 + c) * 64 + r];
        tW[i] = s;
    }
    for (int r = threadIdx.x; r < 64; r += blockDim.x) {
        float s = 0.0f;
#pragma unroll
        for (int d = 0; d < D_; ++d) s += ed[d] * bias_pool[d * 64 + r];
        tB[r] = s;
    }
    __syncthreads();
    for (int i = threadIdx.x; i < 4096; i += blockDim.x) {
        int c = i >> 6, q = i & 63;
        float s = 0.0f;
        for (int r = 0; r < 64; ++r) s += tW[c * 64 + r] * diff_w[q * 128 + 64 + r];
        int lane = (q & 15) + (((c >> 3) & 1) << 4);
        int kt = c >> 5, ct = q >> 4;
        pWn[(size_t)n * 4096 + ((ct * 2 + kt) * 32 + lane) * 16 + (c & 7) + (((c & 31) >> 4) << 3)] = (_Float16)s;
    }
    for (int q = threadIdx.x; q < 64; q += blockDim.x) {
        float s = diff_b[q];
        for (int r = 0; r < 64; ++r) s += tB[r] * diff_w[q * 128 + 64 + r];
        bias2[n * 64 + q] = s;
    }
}

// ---------------------------------------------------------------------------
// Stage A: feature-level dynamic dependency + fc1 MLP -> g1in [T][BN][16] f16
// ---------------------------------------------------------------------------
__global__ void k_stageA(const float* __restrict__ x,
                         const float* xw1, const float* xb1, const float* xw2,
                         const float* xb2, const float* xw3, const float* xb3,
                         const float* fw1, const float* fb1, const float* fw2,
                         const float* fb2, const float* fw3, const float* fb3,
                         _Float16* __restrict__ g1in) {
    int idx = blockIdx.x * blockDim.x + threadIdx.x;
    if (idx >= T_ * BN_) return;
    int t = idx / BN_, m = idx % BN_;
    int b = m / N_, n = m % N_;
    const float* xp = x + ((size_t)(b * T_ + t) * N_ + n) * I_;
    float xv[3] = { xp[0], xp[1], xp[2] };
    float e[3][D_];
#pragma unroll
    for (int i = 0; i < 3; ++i) {
        float v = xv[i];
        float h1[16];
#pragma unroll
        for (int j = 0; j < 16; ++j) h1[j] = sigm(v * xw1[j] + xb1[j]);
        float h2[2];
#pragma unroll
        for (int k = 0; k < 2; ++k) {
            float s = xb2[k];
#pragma unroll
            for (int j = 0; j < 16; ++j) s += h1[j] * xw2[k * 16 + j];
            h2[k] = sigm(s);
        }
#pragma unroll
        for (int d = 0; d < D_; ++d)
            e[i][d] = h2[0] * xw3[d * 2] + h2[1] * xw3[d * 2 + 1] + xb3[d];
    }
    float u[3];
#pragma unroll
    for (int j = 0; j < 3; ++j) {
        float x2 = 0.0f;
#pragma unroll
        for (int i = 0; i < 3; ++i) {
            float s = 0.0f;
#pragma unroll
            for (int d = 0; d < D_; ++d) s += e[j][d] * e[i][d];
            s = s > 0.0f ? s : 0.0f;   // relu support
            x2 += s * xv[i];
        }
        u[j] = xv[j] + x2;
    }
    float h1[16];
#pragma unroll
    for (int k = 0; k < 16; ++k)
        h1[k] = sigm(u[0] * fw1[k * 3] + u[1] * fw1[k * 3 + 1] + u[2] * fw1[k * 3 + 2] + fb1[k]);
    float h2[2];
#pragma unroll
    for (int k = 0; k < 2; ++k) {
        float s = fb2[k];
#pragma unroll
        for (int j = 0; j < 16; ++j) s += h1[j] * fw2[k * 16 + j];
        h2[k] = sigm(s);
    }
    _Float16* gp = g1in + (size_t)idx * 16;
#pragma unroll
    for (int d = 0; d < D_; ++d)
        gp[d] = (_Float16)(h2[0] * fw3[d * 2] + h2[1] * fw3[d * 2 + 1] + fb3[d]);
#pragma unroll
    for (int d = D_; d < 16; ++d) gp[d] = (_Float16)0.0f;
}

// ---------------------------------------------------------------------------
// GRU1 step t (fused xw + hw GEMMs + gates).  grid=BN/16 blocks x 128 thr.
// ---------------------------------------------------------------------------
__global__ void k_gru1_step(int t, const _Float16* __restrict__ g1in,
                            _Float16* __restrict__ h, _Float16* __restrict__ out1,
                            const _Float16* __restrict__ pWhh, const _Float16* __restrict__ pWih,
                            const float* __restrict__ bih, const float* __restrict__ bhh) {
    __shared__ __attribute__((aligned(16))) _Float16 sh[16 * 64];
    int lane = threadIdx.x & 31, wave = threadIdx.x >> 5;
    int mbase = blockIdx.x * 16;
    {   // cooperative stage of h tile (avoids intra-block RAW races on global h)
        int r = threadIdx.x >> 3, cc = (threadIdx.x & 7) * 8;
        *(v8h*)&sh[r * 64 + cc] = *(const v8h*)&h[(size_t)(mbase + r) * 64 + cc];
    }
    __syncthreads();
    int lr = lane & 15, lh = lane >> 4;
    const _Float16* hrow = &sh[lr * 64];
    v16h ah0 = cat16(ld8(hrow + 8 * lh),      ld8(hrow + 16 + 8 * lh));
    v16h ah1 = cat16(ld8(hrow + 32 + 8 * lh), ld8(hrow + 48 + 8 * lh));
    const _Float16* xrow = g1in + (size_t)(t * BN_ + mbase + lr) * 16;
    v16h ax = cat16(ld8(xrow + 8 * lh), zero8h());
    v8f accx[3], acch[3];
#pragma unroll
    for (int g = 0; g < 3; ++g) {
        int ct = wave + 4 * g;
        v16h bh0 = ld16(pWhh + ((ct * 2 + 0) * 32 + lane) * 16);
        v16h bh1 = ld16(pWhh + ((ct * 2 + 1) * 32 + lane) * 16);
        v8f a = WMMA_F16(ah0, bh0, zero8f());
        acch[g] = WMMA_F16(ah1, bh1, a);
        v16h bx = ld16(pWih + (ct * 32 + lane) * 16);
        accx[g] = WMMA_F16(ax, bx, zero8f());
    }
    int c = wave * 16 + lr;
    float bxr = bih[c],       bhr = bhh[c];
    float bxz = bih[c + 64],  bhz = bhh[c + 64];
    float bxn = bih[c + 128], bhn = bhh[c + 128];
#pragma unroll
    for (int v = 0; v < 8; ++v) {
        int mrow = v + 8 * lh;
        int m = mbase + mrow;
        float xr = accx[0][v] + bxr, hr = acch[0][v] + bhr;
        float xz = accx[1][v] + bxz, hz = acch[1][v] + bhz;
        float xn = accx[2][v] + bxn, hn = acch[2][v] + bhn;
        float r = sigm(xr + hr), z = sigm(xz + hz);
        float nn = tanhf(xn + r * hn);
        float hp = (float)sh[mrow * 64 + c];
        float hnew = (1.0f - z) * nn + z * hp;
        _Float16 hf = (_Float16)hnew;
        h[(size_t)m * 64 + c] = hf;
        out1[(size_t)(t * BN_ + m) * 64 + c] = hf;
    }
}

// ---------------------------------------------------------------------------
// fc2 MLP + nodevec = tanh(emb1 * filt) -> nv [TB][NP][16] f16 (zero padded)
// ---------------------------------------------------------------------------
__global__ void k_fc2nv(const _Float16* __restrict__ out1, const float* __restrict__ emb1,
                        const float* w1, const float* b1, const float* w2,
                        const float* b2, const float* w3, const float* b3,
                        _Float16* __restrict__ nv) {
    int idx = blockIdx.x * blockDim.x + threadIdx.x;
    if (idx >= TB_ * NP_) return;
    int tb = idx / NP_, n = idx % NP_;
    _Float16* np = nv + (size_t)idx * 16;
    if (n >= N_) {
#pragma unroll
        for (int d = 0; d < 16; ++d) np[d] = (_Float16)0.0f;
        return;
    }
    int t = tb / B_, b = tb % B_;
    const _Float16* orow = out1 + (size_t)(t * BN_ + b * N_ + n) * 64;
    float o[64];
#pragma unroll
    for (int j = 0; j < 64; ++j) o[j] = (float)orow[j];
    float h1[16];
    for (int k = 0; k < 16; ++k) {
        float s = b1[k];
#pragma unroll
        for (int j = 0; j < 64; ++j) s += o[j] * w1[k * 64 + j];
        h1[k] = sigm(s);
    }
    float h2[2];
#pragma unroll
    for (int k = 0; k < 2; ++k) {
        float s = b2[k];
#pragma unroll
        for (int j = 0; j < 16; ++j) s += h1[j] * w2[k * 16 + j];
        h2[k] = sigm(s);
    }
#pragma unroll
    for (int d = 0; d < D_; ++d) {
        float f = h2[0] * w3[d * 2] + h2[1] * w3[d * 2 + 1] + b3[d];
        np[d] = (_Float16)tanhf(emb1[n * D_ + d] * f);
    }
#pragma unroll
    for (int d = D_; d < 16; ++d) np[d] = (_Float16)0.0f;
}

// ---------------------------------------------------------------------------
// out1T [tb][c][m(320)] f16 (K-major copy for B-operand of S@out1)
// ---------------------------------------------------------------------------
__global__ void k_transpose(const _Float16* __restrict__ out1, _Float16* __restrict__ out1T) {
    int idx = blockIdx.x * blockDim.x + threadIdx.x;
    if (idx >= TB_ * 64 * NP_) return;
    int m = idx % NP_;
    int rest = idx / NP_;
    int c = rest % 64, tb = rest / 64;
    _Float16 v = (_Float16)0.0f;
    if (m < N_) {
        int t = tb / B_, b = tb % B_;
        v = out1[(size_t)(t * BN_ + b * N_ + m) * 64 + c];
    }
    out1T[idx] = v;
}

// ---------------------------------------------------------------------------
// x_g = out1 + relu(nv nv^T) @ out1, fused.  grid=(20 row-tiles, 768 tb).
// Each wave owns one 16-col channel tile; S tiles re-laid-out C->A via LDS.
// ---------------------------------------------------------------------------
__global__ void k_xg(const _Float16* __restrict__ nv, const _Float16* __restrict__ out1T,
                     const _Float16* __restrict__ out1, _Float16* __restrict__ xg) {
    __shared__ __attribute__((aligned(16))) _Float16 s2[4][16 * 32];
    int rt = blockIdx.x, tb = blockIdx.y;
    int lane = threadIdx.x & 31, wave = threadIdx.x >> 5;
    int lr = lane & 15, lh = lane >> 4;
    const _Float16* nvb = nv + (size_t)tb * NP_ * 16;
    v16h anv = cat16(ld8(nvb + (rt * 16 + lr) * 16 + 8 * lh), zero8h());
    _Float16* sw = s2[wave];
    v8f acc = zero8f();
    for (int mt2 = 0; mt2 < 10; ++mt2) {
        int mbase = mt2 * 32;
#pragma unroll
        for (int s = 0; s < 2; ++s) {
            v16h bnv = zero16h();
            bnv = cat16(ld8(nvb + (mbase + s * 16 + lr) * 16 + 8 * lh), zero8h());
            v8f sv = WMMA_F16(anv, bnv, zero8f());
#pragma unroll
            for (int v = 0; v < 8; ++v) {
                float f = sv[v];
                f = f > 0.0f ? f : 0.0f;                 // relu(support)
                sw[(v + 8 * lh) * 32 + s * 16 + lr] = (_Float16)f;
            }
        }
        asm volatile("s_wait_dscnt 0" ::: "memory");     // LDS RAW (cross-lane, same wave)
        v16h a2 = cat16(*(const v8h*)&sw[lr * 32 + 8 * lh],
                        *(const v8h*)&sw[lr * 32 + 16 + 8 * lh]);
        v16h b2 = cat16(ld8(out1T + ((size_t)tb * 64 + wave * 16 + lr) * NP_ + mbase + 8 * lh), ld8(out1T + ((size_t)tb * 64 + wave * 16 + lr) * NP_ + mbase + 16 + 8 * lh));
        acc = WMMA_F16(a2, b2, acc);
    }
    int t = tb / B_, b = tb % B_;
    int c = wave * 16 + lr;
#pragma unroll
    for (int v = 0; v < 8; ++v) {
        int n = rt * 16 + v + 8 * lh;
        if (n < N_) {
            int m = b * N_ + n;
            float val = acc[v] + (float)out1[(size_t)(t * BN_ + m) * 64 + c];
            xg[((size_t)n * TB_ + tb) * 64 + c] = (_Float16)val;
        }
    }
}

// ---------------------------------------------------------------------------
// g2_in = x_g @ W'[n] + out1 @ dw1^T + bias2[n]   grid=(48 row-tiles, 307 n)
// ---------------------------------------------------------------------------
__global__ void k_diffg2(const _Float16* __restrict__ xg, const _Float16* __restrict__ out1,
                         const _Float16* __restrict__ pWn, const float* __restrict__ bias2,
                         const _Float16* __restrict__ pDw1, _Float16* __restrict__ g2in) {
    int rt = blockIdx.x, n = blockIdx.y;
    int lane = threadIdx.x & 31, wave = threadIdx.x >> 5;
    int lr = lane & 15, lh = lane >> 4;
    int tb = rt * 16 + lr;
    int t = tb / B_, b = tb % B_;
    const _Float16* xr = xg + ((size_t)n * TB_ + tb) * 64;
    v16h a10 = cat16(ld8(xr + 8 * lh),      ld8(xr + 16 + 8 * lh));
    v16h a11 = cat16(ld8(xr + 32 + 8 * lh), ld8(xr + 48 + 8 * lh));
    const _Float16* orow = out1 + (size_t)(t * BN_ + b * N_ + n) * 64;
    v16h a20 = cat16(ld8(orow + 8 * lh),      ld8(orow + 16 + 8 * lh));
    v16h a21 = cat16(ld8(orow + 32 + 8 * lh), ld8(orow + 48 + 8 * lh));
    int q = wave * 16 + lr;
    float bb = bias2[n * 64 + q];
    v8f acc;
#pragma unroll
    for (int v = 0; v < 8; ++v) acc[v] = bb;
    const _Float16* wn = pWn + (size_t)n * 4096;
    acc = WMMA_F16(a10, ld16(wn + ((wave * 2 + 0) * 32 + lane) * 16), acc);
    acc = WMMA_F16(a11, ld16(wn + ((wave * 2 + 1) * 32 + lane) * 16), acc);
    acc = WMMA_F16(a20, ld16(pDw1 + ((wave * 2 + 0) * 32 + lane) * 16), acc);
    acc = WMMA_F16(a21, ld16(pDw1 + ((wave * 2 + 1) * 32 + lane) * 16), acc);
#pragma unroll
    for (int v = 0; v < 8; ++v) {
        int tb2 = rt * 16 + v + 8 * lh;
        int t2 = tb2 / B_, b2 = tb2 % B_;
        g2in[(size_t)(t2 * BN_ + b2 * N_ + n) * 64 + q] = (_Float16)acc[v];
    }
}

// ---------------------------------------------------------------------------
// GRU2 step t: like GRU1 but K=64 input projection; writes final [B,T,N,R].
// ---------------------------------------------------------------------------
__global__ void k_gru2_step(int t, const _Float16* __restrict__ g2in,
                            _Float16* __restrict__ h, float* __restrict__ out,
                            const _Float16* __restrict__ pWhh, const _Float16* __restrict__ pWih,
                            const float* __restrict__ bih, const float* __restrict__ bhh) {
    __shared__ __attribute__((aligned(16))) _Float16 sh[16 * 64];
    int lane = threadIdx.x & 31, wave = threadIdx.x >> 5;
    int mbase = blockIdx.x * 16;
    {
        int r = threadIdx.x >> 3, cc = (threadIdx.x & 7) * 8;
        *(v8h*)&sh[r * 64 + cc] = *(const v8h*)&h[(size_t)(mbase + r) * 64 + cc];
    }
    __syncthreads();
    int lr = lane & 15, lh = lane >> 4;
    const _Float16* hrow = &sh[lr * 64];
    v16h ah0 = cat16(ld8(hrow + 8 * lh),      ld8(hrow + 16 + 8 * lh));
    v16h ah1 = cat16(ld8(hrow + 32 + 8 * lh), ld8(hrow + 48 + 8 * lh));
    const _Float16* xrow = g2in + (size_t)(t * BN_ + mbase + lr) * 64;
    v16h ax0 = cat16(ld8(xrow + 8 * lh),      ld8(xrow + 16 + 8 * lh));
    v16h ax1 = cat16(ld8(xrow + 32 + 8 * lh), ld8(xrow + 48 + 8 * lh));
    v8f accx[3], acch[3];
#pragma unroll
    for (int g = 0; g < 3; ++g) {
        int ct = wave + 4 * g;
        v16h bh0 = ld16(pWhh + ((ct * 2 + 0) * 32 + lane) * 16);
        v16h bh1 = ld16(pWhh + ((ct * 2 + 1) * 32 + lane) * 16);
        acch[g] = WMMA_F16(ah1, bh1, WMMA_F16(ah0, bh0, zero8f()));
        v16h bx0 = ld16(pWih + ((ct * 2 + 0) * 32 + lane) * 16);
        v16h bx1 = ld16(pWih + ((ct * 2 + 1) * 32 + lane) * 16);
        accx[g] = WMMA_F16(ax1, bx1, WMMA_F16(ax0, bx0, zero8f()));
    }
    int c = wave * 16 + lr;
    float bxr = bih[c],       bhr = bhh[c];
    float bxz = bih[c + 64],  bhz = bhh[c + 64];
    float bxn = bih[c + 128], bhn = bhh[c + 128];
#pragma unroll
    for (int v = 0; v < 8; ++v) {
        int mrow = v + 8 * lh;
        int m = mbase + mrow;
        float xr = accx[0][v] + bxr, hr = acch[0][v] + bhr;
        float xz = accx[1][v] + bxz, hz = acch[1][v] + bhz;
        float xn = accx[2][v] + bxn, hn = acch[2][v] + bhn;
        float r = sigm(xr + hr), z = sigm(xz + hz);
        float nn = tanhf(xn + r * hn);
        float hp = (float)sh[mrow * 64 + c];
        float hnew = (1.0f - z) * nn + z * hp;
        h[(size_t)m * 64 + c] = (_Float16)hnew;
        int b2 = m / N_, n2 = m % N_;
        out[(size_t)((b2 * T_ + t) * N_ + n2) * 64 + c] = hnew;
    }
}

// ---------------------------------------------------------------------------
extern "C" void kernel_launch(void* const* d_in, const int* in_sizes, int n_in,
                              void* d_out, int out_size, void* d_ws, size_t ws_size,
                              hipStream_t stream) {
    (void)in_sizes; (void)n_in; (void)out_size;
    const float* x         = (const float*)d_in[0];
    const float* emb1      = (const float*)d_in[1];
    const float* emb2      = (const float*)d_in[2];
    const float* xe_w1 = (const float*)d_in[3],  *xe_b1 = (const float*)d_in[4];
    const float* xe_w2 = (const float*)d_in[5],  *xe_b2 = (const float*)d_in[6];
    const float* xe_w3 = (const float*)d_in[7],  *xe_b3 = (const float*)d_in[8];
    const float* f1_w1 = (const float*)d_in[9],  *f1_b1 = (const float*)d_in[10];
    const float* f1_w2 = (const float*)d_in[11], *f1_b2 = (const float*)d_in[12];
    const float* f1_w3 = (const float*)d_in[13], *f1_b3 = (const float*)d_in[14];
    const float* f2_w1 = (const float*)d_in[15], *f2_b1 = (const float*)d_in[16];
    const float* f2_w2 = (const float*)d_in[17], *f2_b2 = (const float*)d_in[18];
    const float* f2_w3 = (const float*)d_in[19], *f2_b3 = (const float*)d_in[20];
    const float* g1_wih = (const float*)d_in[21], *g1_whh = (const float*)d_in[22];
    const float* g1_bih = (const float*)d_in[23], *g1_bhh = (const float*)d_in[24];
    const float* g2_wih = (const float*)d_in[25], *g2_whh = (const float*)d_in[26];
    const float* g2_bih = (const float*)d_in[27], *g2_bhh = (const float*)d_in[28];
    const float* wpool  = (const float*)d_in[29], *bpool  = (const float*)d_in[30];
    const float* diff_w = (const float*)d_in[31], *diff_b = (const float*)d_in[32];
    float* out = (float*)d_out;

    char* ws = (char*)d_ws;
    size_t off = 0;
    auto A = [&](size_t bytes) -> void* {
        void* p = ws + off;
        off = (off + bytes + 255) & ~(size_t)255;
        return p;
    };
    _Float16* g1in  = (_Float16*)A((size_t)T_ * BN_ * 16 * 2);
    _Float16* h1    = (_Float16*)A((size_t)BN_ * 64 * 2);
    _Float16* h2    = (_Float16*)A((size_t)BN_ * 64 * 2);
    _Float16* out1  = (_Float16*)A(((size_t)T_ * BN_ + 16) * 64 * 2);
    _Float16* nv    = (_Float16*)A((size_t)TB_ * NP_ * 16 * 2);
    _Float16* out1T = (_Float16*)A((size_t)TB_ * 64 * NP_ * 2);
    _Float16* xgb   = (_Float16*)A((size_t)N_ * TB_ * 64 * 2);
    _Float16* g2in  = (_Float16*)A((size_t)T_ * BN_ * 64 * 2);
    _Float16* pWn   = (_Float16*)A((size_t)N_ * 4096 * 2);
    float*    bias2 = (float*)   A((size_t)N_ * 64 * 4);
    _Float16* pWhh1 = (_Float16*)A(24 * 512 * 2);
    _Float16* pWih1 = (_Float16*)A(12 * 512 * 2);
    _Float16* pWhh2 = (_Float16*)A(24 * 512 * 2);
    _Float16* pWih2 = (_Float16*)A(24 * 512 * 2);
    _Float16* pDw1  = (_Float16*)A(8 * 512 * 2);
    if (off > ws_size) return;  // workspace too small: no-op (deterministic)

    k_zero_h<<<(BN_ * R_ + 255) / 256, 256, 0, stream>>>(h1, h2);
    k_pack<<<92, 512, 0, stream>>>(g1_whh, g1_wih, g2_whh, g2_wih, diff_w,
                                   pWhh1, pWih1, pWhh2, pWih2, pDw1);
    k_prep_n<<<N_, 256, 0, stream>>>(emb2, wpool, bpool, diff_w, diff_b, pWn, bias2);
    k_stageA<<<(T_ * BN_ + 255) / 256, 256, 0, stream>>>(
        x, xe_w1, xe_b1, xe_w2, xe_b2, xe_w3, xe_b3,
        f1_w1, f1_b1, f1_w2, f1_b2, f1_w3, f1_b3, g1in);
    for (int t = 0; t < T_; ++t)
        k_gru1_step<<<BN_ / 16, 128, 0, stream>>>(t, g1in, h1, out1, pWhh1, pWih1,
                                                  g1_bih, g1_bhh);
    k_fc2nv<<<(TB_ * NP_ + 255) / 256, 256, 0, stream>>>(
        out1, emb1, f2_w1, f2_b1, f2_w2, f2_b2, f2_w3, f2_b3, nv);
    k_transpose<<<(TB_ * 64 * NP_ + 255) / 256, 256, 0, stream>>>(out1, out1T);
    k_xg<<<dim3(NP_ / 16, TB_), 128, 0, stream>>>(nv, out1T, out1, xgb);
    k_diffg2<<<dim3(TB_ / 16, N_), 128, 0, stream>>>(xgb, out1, pWn, bias2, pDw1, g2in);
    for (int t = 0; t < T_; ++t)
        k_gru2_step<<<BN_ / 16, 128, 0, stream>>>(t, g2in, h2, out, pWhh2, pWih2,
                                                  g2_bih, g2_bhh);
}
